// MambaWrap_12807592476900
// MI455X (gfx1250) — matrix operand order repaired and checked by reading, not compiled, not run on hardware
//
#include <hip/hip_runtime.h>
#include <stdint.h>

// ---------------- problem constants ----------------
#define D_MODEL   1024
#define D_INNER   2048
#define D_STATE   128
#define NHEADS    16
#define HEADDIM   128
#define D_CONV    4
#define D_XBC     (D_INNER + 2 * D_STATE)               // 2304
#define D_IN_PROJ (2 * D_INNER + 2 * D_STATE + NHEADS)  // 4368
#define D_IN_PAD  4480                                   // padded to multiple of 128
#define BATCH     2
#define SEQLEN    2048
#define TOKENS    (BATCH * SEQLEN)                       // 4096
#define EPS       1e-5f

typedef _Float16 half_t;
typedef __attribute__((ext_vector_type(16))) _Float16 v16h;
typedef __attribute__((ext_vector_type(8)))  _Float16 v8h;
typedef __attribute__((ext_vector_type(8)))  float    v8f;

union V16 {
    v16h v;
    v8h  h[2];
};
union H2 {
    _Float16 h[2];
    uint32_t u;
};

// ---------------- small helpers ----------------
__device__ inline float block_reduce_sum(float v, float* sbuf) {
    int tid = threadIdx.x;
    sbuf[tid] = v;
    __syncthreads();
    for (int s = blockDim.x >> 1; s > 0; s >>= 1) {
        if (tid < s) sbuf[tid] += sbuf[tid + s];
        __syncthreads();
    }
    float r = sbuf[0];
    __syncthreads();
    return r;
}

__device__ inline float silu_f(float x) { return x / (1.0f + __expf(-x)); }

// ---------------- LayerNorm -> fp16 activations ----------------
__global__ void ln_kernel(const float* __restrict__ x,
                          const float* __restrict__ w,
                          const float* __restrict__ b,
                          half_t* __restrict__ out) {
    __shared__ float sbuf[256];
    int token = blockIdx.x;
    int tid = threadIdx.x;
    const float* xr = x + (size_t)token * D_MODEL;
    float v[4];
    float s = 0.f, ss = 0.f;
#pragma unroll
    for (int i = 0; i < 4; ++i) {
        v[i] = xr[tid + i * 256];
        s += v[i];
        ss += v[i] * v[i];
    }
    float total  = block_reduce_sum(s, sbuf);
    float total2 = block_reduce_sum(ss, sbuf);
    float mean = total * (1.0f / D_MODEL);
    float var  = total2 * (1.0f / D_MODEL) - mean * mean;
    float rstd = rsqrtf(var + EPS);
#pragma unroll
    for (int i = 0; i < 4; ++i) {
        int c = tid + i * 256;
        out[(size_t)token * D_MODEL + c] = (half_t)((v[i] - mean) * rstd * w[c] + b[c]);
    }
}

// ---------------- weight fp32 -> fp16 converters ----------------
__global__ void cvt_win_kernel(const float* __restrict__ W, half_t* __restrict__ o) {
    const size_t total = (size_t)D_MODEL * D_IN_PAD;
    for (size_t i = (size_t)blockIdx.x * blockDim.x + threadIdx.x; i < total;
         i += (size_t)gridDim.x * blockDim.x) {
        size_t r = i / D_IN_PAD;
        int col = (int)(i % D_IN_PAD);
        o[i] = (col < D_IN_PROJ) ? (half_t)W[r * D_IN_PROJ + col] : (half_t)0.0f;
    }
}

__global__ void cvt_wout_kernel(const float* __restrict__ W, half_t* __restrict__ o) {
    const size_t total = (size_t)D_INNER * D_MODEL;
    for (size_t i = (size_t)blockIdx.x * blockDim.x + threadIdx.x; i < total;
         i += (size_t)gridDim.x * blockDim.x) {
        o[i] = (half_t)W[i];
    }
}

// ---------------- WMMA GEMM: C[M,N] = A[M,K] * B[K,N] (+resid) ----------------
// block: 256 threads = 8 waves (4 in M x 2 in N); block tile 128x128; K step 32.
// As: row-major M x K tile, row stride 40 halves (80 B, 16B aligned).
// Bsw: TRANSPOSED tile [col][k], row stride 40 halves -> every WMMA B fragment
//      is two contiguous 16B chunks (ds_load_b128), no scalar gathers.
#define LDA_S 40
#define LDB_SW 40

template <bool ADD_RESID>
__global__ void gemm_wmma_kernel(const half_t* __restrict__ A,
                                 const half_t* __restrict__ B,
                                 float* __restrict__ C,
                                 const float* __restrict__ resid,
                                 int M, int N, int K, int ldc) {
    __shared__ half_t As[128 * LDA_S];
    __shared__ half_t Bsw[128 * LDB_SW];

    const int tid    = threadIdx.x;
    const int lane   = tid & 31;
    const int waveId = tid >> 5;
    const int wm     = waveId & 3;   // 0..3 : 32-row slice
    const int wn     = waveId >> 2;  // 0..1 : 64-col slice
    const int blockM = blockIdx.y * 128;
    const int blockN = blockIdx.x * 128;

    v8f acc[2][4];
#pragma unroll
    for (int mi = 0; mi < 2; ++mi)
#pragma unroll
        for (int ni = 0; ni < 4; ++ni) acc[mi][ni] = (v8f){};

    // A staging coordinates: thread copies 16 halves (32 B) of one row
    const int aRow = tid >> 1;            // 0..127
    const int aCh  = (tid & 1) * 16;      // 0 / 16 halves
    // B staging coordinates: thread transposes a 2(k) x 8(col) patch
    const int colchunk = tid & 15;        // 8-column chunk
    const int kpair    = tid >> 4;        // 0..15 -> k = 2*kpair

    for (int k0 = 0; k0 < K; k0 += 32) {
        // --- stage A tile 128x32 : global b128 -> LDS b128 ---
        {
            const uint4* src = (const uint4*)(A + (size_t)(blockM + aRow) * K + k0 + aCh);
            uint4* dst = (uint4*)(&As[aRow * LDA_S + aCh]);
            dst[0] = src[0];
            dst[1] = src[1];
        }
        // --- stage B tile 32x128 transposed: 2 x global b128, pack pairs, 8 x ds_store_b32 ---
        {
            const half_t* g0 = B + (size_t)(k0 + 2 * kpair) * N + blockN + colchunk * 8;
            v8h r0 = *(const v8h*)g0;
            v8h r1 = *(const v8h*)(g0 + N);
#pragma unroll
            for (int j = 0; j < 8; ++j) {
                H2 p;
                p.h[0] = r0[j];
                p.h[1] = r1[j];
                *(uint32_t*)&Bsw[(colchunk * 8 + j) * LDB_SW + 2 * kpair] = p.u;
            }
        }
        // prefetch next K tile (emits global_prefetch_b8)
        if (k0 + 32 < K) {
            __builtin_prefetch(A + (size_t)(blockM + aRow) * K + k0 + 32 + aCh, 0, 1);
            __builtin_prefetch(B + (size_t)(k0 + 32 + 2 * kpair) * N + blockN + colchunk * 8, 0, 1);
        }
        __syncthreads();

        // --- A fragments: ISA 16-bit A 16x32 layout; two aligned b128 LDS loads ---
        v16h afrag[2];
        {
            const int kb = (lane < 16) ? 0 : 8;
#pragma unroll
            for (int mi = 0; mi < 2; ++mi) {
                const half_t* r = &As[(wm * 32 + mi * 16 + (lane & 15)) * LDA_S];
                V16 u;
                u.h[0] = *(const v8h*)(r + kb);
                u.h[1] = *(const v8h*)(r + kb + 16);
                afrag[mi] = u.v;
            }
        }
        // --- B fragments: lane n = column; elements 0-7 = K rows kb..kb+7, 8-15 = kb+16..kb+23; contiguous in Bsw ---
        v16h bfrag[4];
        {
            const int kb = (lane < 16) ? 0 : 8;
#pragma unroll
            for (int ni = 0; ni < 4; ++ni) {
                const half_t* c = &Bsw[(wn * 64 + ni * 16 + (lane & 15)) * LDB_SW + kb];
                V16 u;
                u.h[0] = *(const v8h*)c;
                u.h[1] = *(const v8h*)(c + 16);
                bfrag[ni] = u.v;
            }
        }
#pragma unroll
        for (int mi = 0; mi < 2; ++mi)
#pragma unroll
            for (int ni = 0; ni < 4; ++ni)
                acc[mi][ni] = __builtin_amdgcn_wmma_f32_16x16x32_f16(
                    false, afrag[mi], false, bfrag[ni], (short)0, acc[mi][ni],
                    false, false);
        __syncthreads();
    }

    // --- epilogue: C/D layout (VGPR r -> M=r lanes0-15, M=8+r lanes16-31) ---
    const int mOfs = (lane >= 16) ? 8 : 0;
    const int cColL = lane & 15;
#pragma unroll
    for (int mi = 0; mi < 2; ++mi) {
#pragma unroll
        for (int ni = 0; ni < 4; ++ni) {
            int colg = blockN + wn * 64 + ni * 16 + cColL;
#pragma unroll
            for (int r = 0; r < 8; ++r) {
                int rowg = blockM + wm * 32 + mi * 16 + mOfs + r;
                float v = acc[mi][ni][r];
                if (ADD_RESID) v += resid[(size_t)rowg * ldc + colg];
                C[(size_t)rowg * ldc + colg] = v;
            }
        }
    }
}

// ---------------- depthwise causal conv (4 taps) + SiLU ----------------
__global__ void conv_silu_kernel(const float* __restrict__ zx,
                                 const float* __restrict__ cw,
                                 const float* __restrict__ cb,
                                 float* __restrict__ out) {
    const size_t total = (size_t)TOKENS * D_XBC;
    for (size_t idx = (size_t)blockIdx.x * blockDim.x + threadIdx.x; idx < total;
         idx += (size_t)gridDim.x * blockDim.x) {
        int c = (int)(idx % D_XBC);
        size_t token = idx / D_XBC;
        int l  = (int)(token % SEQLEN);
        int bb = (int)(token / SEQLEN);
        float acc = cb[c];
#pragma unroll
        for (int k = 0; k < D_CONV; ++k) {
            int ls = l + k - (D_CONV - 1);
            if (ls >= 0)
                acc += zx[((size_t)bb * SEQLEN + ls) * D_IN_PAD + D_INNER + c] *
                       cw[c * D_CONV + k];
        }
        out[idx] = silu_f(acc);
    }
}

// ---------------- dt = softplus(raw+bias), dA = exp(dt*A) ----------------
__global__ void dtda_kernel(const float* __restrict__ zx,
                            const float* __restrict__ dt_bias,
                            const float* __restrict__ A_log,
                            float* __restrict__ dt, float* __restrict__ dA) {
    const size_t total = (size_t)TOKENS * NHEADS;
    for (size_t idx = (size_t)blockIdx.x * blockDim.x + threadIdx.x; idx < total;
         idx += (size_t)gridDim.x * blockDim.x) {
        int h = (int)(idx % NHEADS);
        size_t token = idx / NHEADS;
        float v = zx[token * D_IN_PAD + (D_IN_PROJ - NHEADS) + h] + dt_bias[h];
        float dtv = (v > 20.0f) ? v : log1pf(__expf(v));
        float Ah = -__expf(A_log[h]);
        dt[idx] = dtv;
        dA[idx] = __expf(dtv * Ah);
    }
}

// ---------------- selective scan: 32 blocks (b,h), 512 threads ----------------
// thread layout: p = tid>>2 (head dim row), nchunk = tid&3 (32 states each).
// State lives in registers (fully unrolled); reduce across the 4 lanes sharing
// a row with wave shuffles.
__global__ void scan_kernel(const float* __restrict__ xbc,
                            const float* __restrict__ dt,
                            const float* __restrict__ dA,
                            const float* __restrict__ Dv,
                            float* __restrict__ y) {
    __shared__ float Bs[D_STATE];
    __shared__ float Cs[D_STATE];
    const int bh = blockIdx.x;
    const int b  = bh >> 4;
    const int h  = bh & 15;
    const int tid = threadIdx.x;
    const int p  = tid >> 2;           // 0..127
    const int nc = tid & 3;            // 0..3
    const int nbase = nc * 32;

    float st[32];
#pragma unroll
    for (int n = 0; n < 32; ++n) st[n] = 0.0f;
    const float Dh = Dv[h];

    for (int t = 0; t < SEQLEN; ++t) {
        const size_t token = (size_t)b * SEQLEN + t;
        const float* row = xbc + token * D_XBC;
        __syncthreads();
        if (tid < 128)          Bs[tid]       = row[D_INNER + tid];
        else if (tid < 256)     Cs[tid - 128] = row[D_INNER + D_STATE + (tid - 128)];
        __syncthreads();

        const float xh  = row[(size_t)h * HEADDIM + p];
        const float dtv = dt[token * NHEADS + h];
        const float da  = dA[token * NHEADS + h];
        const float coef = dtv * xh;

        float acc = 0.0f;
#pragma unroll
        for (int n = 0; n < 32; ++n) {
            st[n] = st[n] * da + coef * Bs[nbase + n];
            acc += st[n] * Cs[nbase + n];
        }
        // reduce across the 4 lanes (xor bits 0,1 of lane) sharing this p
        acc += __shfl_xor(acc, 1, 32);
        acc += __shfl_xor(acc, 2, 32);
        if (nc == 0)
            y[token * D_INNER + (size_t)h * HEADDIM + p] = acc + Dh * xh;
    }
}

// ---------------- gate with silu(z) + RMSNorm -> fp16 ----------------
__global__ void gate_rms_kernel(const float* __restrict__ y,
                                const float* __restrict__ zx,
                                const float* __restrict__ rw,
                                half_t* __restrict__ out) {
    __shared__ float sbuf[256];
    int token = blockIdx.x;
    int tid = threadIdx.x;
    float g[8];
    float ss = 0.f;
#pragma unroll
    for (int i = 0; i < 8; ++i) {
        int c = tid + i * 256;
        float yv = y[(size_t)token * D_INNER + c];
        float zv = zx[(size_t)token * D_IN_PAD + c];   // z = first D_INNER cols
        float gv = yv * silu_f(zv);
        g[i] = gv;
        ss += gv * gv;
    }
    float total = block_reduce_sum(ss, sbuf);
    float r = rsqrtf(total * (1.0f / D_INNER) + EPS);
#pragma unroll
    for (int i = 0; i < 8; ++i) {
        int c = tid + i * 256;
        out[(size_t)token * D_INNER + c] = (half_t)(g[i] * r * rw[c]);
    }
}

// ---------------- host driver ----------------
extern "C" void kernel_launch(void* const* d_in, const int* in_sizes, int n_in,
                              void* d_out, int out_size, void* d_ws, size_t ws_size,
                              hipStream_t stream) {
    (void)in_sizes; (void)n_in; (void)out_size; (void)ws_size;
    const float* x       = (const float*)d_in[0];
    const float* ln_w    = (const float*)d_in[1];
    const float* ln_b    = (const float*)d_in[2];
    const float* W_in    = (const float*)d_in[3];
    const float* conv_w  = (const float*)d_in[4];
    const float* conv_b  = (const float*)d_in[5];
    const float* dt_bias = (const float*)d_in[6];
    const float* A_log   = (const float*)d_in[7];
    const float* Dp      = (const float*)d_in[8];
    const float* rms_w   = (const float*)d_in[9];
    const float* W_out   = (const float*)d_in[10];
    float* out = (float*)d_out;

    // workspace carve-up (256B aligned)
    char* ws = (char*)d_ws;
    size_t off = 0;
    auto carve = [&](size_t bytes) -> char* {
        char* p = ws + off;
        off += (bytes + 255) & ~(size_t)255;
        return p;
    };
    half_t* xnH   = (half_t*)carve((size_t)TOKENS * D_MODEL * sizeof(half_t));
    half_t* WinH  = (half_t*)carve((size_t)D_MODEL * D_IN_PAD * sizeof(half_t));
    float*  zx    = (float*) carve((size_t)TOKENS * D_IN_PAD * sizeof(float));
    float*  xbc   = (float*) carve((size_t)TOKENS * D_XBC * sizeof(float));
    float*  dt    = (float*) carve((size_t)TOKENS * NHEADS * sizeof(float));
    float*  dA    = (float*) carve((size_t)TOKENS * NHEADS * sizeof(float));
    float*  yscan = (float*) carve((size_t)TOKENS * D_INNER * sizeof(float));
    half_t* ynH   = (half_t*)carve((size_t)TOKENS * D_INNER * sizeof(half_t));
    half_t* WoutH = (half_t*)carve((size_t)D_INNER * D_MODEL * sizeof(half_t));

    // 1) LayerNorm -> fp16 A matrix
    ln_kernel<<<TOKENS, 256, 0, stream>>>(x, ln_w, ln_b, xnH);
    // 2) weight conversions
    cvt_win_kernel<<<2048, 256, 0, stream>>>(W_in, WinH);
    cvt_wout_kernel<<<2048, 256, 0, stream>>>(W_out, WoutH);
    // 3) in-projection GEMM (WMMA), output padded N = 4480
    gemm_wmma_kernel<false><<<dim3(D_IN_PAD / 128, TOKENS / 128), 256, 0, stream>>>(
        xnH, WinH, zx, nullptr, TOKENS, D_IN_PAD, D_MODEL, D_IN_PAD);
    // 4) depthwise conv + SiLU ; dt / dA precompute
    conv_silu_kernel<<<4096, 256, 0, stream>>>(zx, conv_w, conv_b, xbc);
    dtda_kernel<<<256, 256, 0, stream>>>(zx, dt_bias, A_log, dt, dA);
    // 5) selective scan (register-resident state, serial in L)
    scan_kernel<<<BATCH * NHEADS, 512, 0, stream>>>(xbc, dt, dA, Dp, yscan);
    // 6) gate + RMSNorm -> fp16 A matrix
    gate_rms_kernel<<<TOKENS, 256, 0, stream>>>(yscan, zx, rms_w, ynH);
    // 7) out-projection GEMM (WMMA) with fused residual add
    gemm_wmma_kernel<true><<<dim3(D_MODEL / 128, TOKENS / 128), 256, 0, stream>>>(
        ynH, WoutH, out, x, TOKENS, D_MODEL, D_INNER, D_MODEL);
}
